// FNL2NO_47425028882755
// MI455X (gfx1250) — matrix operand order repaired and checked by reading, not compiled, not run on hardware
//
#include <hip/hip_runtime.h>
#include <cstdint>

// ---------------- problem constants ----------------
#define B_SZ   256
#define T_SZ   32
#define A_IN   128
#define V_IN   256
#define TX_IN  768
#define H_SZ   256
#define G4H    1024          // 4*H
#define NW_L1  8
#define TW0    4
#define R_RK   16
#define O0_SZ  128
#define O1_SZ  64
#define D1_SZ  3073
#define D2_SZ  1025
#define CAT_F  768           // 3*H

typedef __bf16 bf16;
typedef __attribute__((ext_vector_type(16))) __bf16 v16bf;
typedef __attribute__((ext_vector_type(8)))  float  v8f;
typedef __attribute__((ext_vector_type(2)))  float  v2f;
typedef unsigned int u32x4 __attribute__((ext_vector_type(4)));
typedef int          i32x8 __attribute__((ext_vector_type(8)));
typedef int          i32x4 __attribute__((ext_vector_type(4)));

#if defined(__has_builtin)
# if __has_builtin(__builtin_amdgcn_tensor_load_to_lds)
#  define HAVE_TDM 1
# endif
#endif
#if __has_include(<hip/amd_detail/amd_gfx1250_TDM.h>)
# define TDM_SIX_ARG 1
#endif

// ---------------- async global->LDS helpers ----------------
__device__ inline void async_ld_b128(unsigned lds_addr, const void* gptr) {
  unsigned long long ga = (unsigned long long)(uintptr_t)gptr;
  asm volatile("global_load_async_to_lds_b128 %0, %1, off" :: "v"(lds_addr), "v"(ga) : "memory");
}
__device__ inline void wait_async0() {
  asm volatile("s_wait_asynccnt 0x0" ::: "memory");
}
__device__ inline void wait_tensor0() {
#if defined(HAVE_TDM)
  __builtin_amdgcn_s_wait_tensorcnt(0);
#else
  asm volatile("s_wait_tensorcnt 0x0" ::: "memory");
#endif
}

// ---------------- WMMA wrappers ----------------
__device__ inline v8f wmma_bf16(v16bf a, v16bf b, v8f c) {
  return __builtin_amdgcn_wmma_f32_16x16x32_bf16(false, a, false, b, (short)0, c, false, false);
}
__device__ inline v8f wmma_f32(v2f a, v2f b, v8f c) {
  return __builtin_amdgcn_wmma_f32_16x16x4_f32(false, a, false, b, (short)0, c, false, false);
}

// ---------------- fragment loaders (wave32 CDNA5 layouts) ----------------
__device__ inline v16bf load_frag_a_bf(const bf16* base, int ld) {
  int lane = threadIdx.x & 31;
  int m  = lane & 15;
  int kb = (lane >> 4) * 8;
  v16bf a;
#pragma unroll
  for (int i = 0; i < 8; ++i) a[i]     = base[m * ld + kb + i];
#pragma unroll
  for (int i = 0; i < 8; ++i) a[8 + i] = base[m * ld + 16 + kb + i];
  return a;
}
__device__ inline v16bf load_frag_b_bf(const bf16* base, int ld) {
  int lane = threadIdx.x & 31;
  int n  = lane & 15;
  int kb = (lane >> 4) * 8;
  v16bf b;
#pragma unroll
  for (int i = 0; i < 16; ++i) b[i] = base[(kb + i + (i >= 8 ? 8 : 0)) * ld + n];
  return b;
}
__device__ inline v2f load_frag_a_f32(const float* base, int ld) {
  int lane = threadIdx.x & 31;
  int m  = lane & 15;
  int kb = (lane >> 4) * 2;
  v2f a;
  a.x = base[m * ld + kb];
  a.y = base[m * ld + kb + 1];
  return a;
}
__device__ inline v2f load_frag_b_f32(const float* base, int ld) {
  int lane = threadIdx.x & 31;
  int n  = lane & 15;
  int kb = (lane >> 4) * 2;
  v2f b;
  b.x = base[kb * ld + n];
  b.y = base[(kb + 1) * ld + n];
  return b;
}

// ---------------- utility kernels ----------------
__global__ void zero_u32_kernel(unsigned int* p, int n) {
  int i = blockIdx.x * blockDim.x + threadIdx.x;
  if (i < n) p[i] = 0u;
}
__global__ void cvt_bf16_kernel(const float* __restrict__ src, bf16* __restrict__ dst, int n) {
  int i = blockIdx.x * blockDim.x + threadIdx.x;
  if (i < n) dst[i] = (bf16)src[i];
}
__global__ void cvt_t_bf16_kernel(const float* __restrict__ src, bf16* __restrict__ dst,
                                  int rows, int cols) {
  int i = blockIdx.x * blockDim.x + threadIdx.x;
  if (i < rows * cols) {
    int r = i / cols, c = i - r * cols;
    dst[c * rows + r] = (bf16)src[i];
  }
}

// ---------------- bf16 WMMA GEMM (async-LDS staged) ----------------
// block tile 128x64, 8 waves (4x2), wave tile 32x32, K step 32.
// M%128==0, N%64==0, K%32==0; all base pointers 256B aligned (workspace carve).
template<bool CIN_BF16, bool OUT_BF16>
__global__ __launch_bounds__(256)
void gemm_bf16_kernel(const bf16* __restrict__ A, int lda,
                      const bf16* __restrict__ Bm, int ldb,
                      const void* __restrict__ Cin, int ldcin,
                      const float* __restrict__ bias0,
                      const float* __restrict__ bias1,
                      void* __restrict__ Cout, int ldcout,
                      int K) {
  __shared__ bf16 As[128][40];   // row stride 80B (16B aligned)
  __shared__ bf16 Bs[32][72];    // row stride 144B (16B aligned)
  const int tid  = threadIdx.x;
  const int lane = tid & 31;
  const int w    = tid >> 5;
  const int wm   = w & 3;
  const int wn   = w >> 2;
  const int m0   = blockIdx.y * 128;
  const int n0   = blockIdx.x * 64;

  v8f acc[2][2] = {};

  for (int kb = 0; kb < K; kb += 32) {
    __syncthreads();
    // A tile 128x32 bf16, 8 bf16 (16B) per async op
#pragma unroll
    for (int i = tid; i < 512; i += 256) {
      int m = i >> 2, ko = (i & 3) * 8;
      async_ld_b128((unsigned)(uintptr_t)&As[m][ko],
                    A + (size_t)(m0 + m) * lda + kb + ko);
    }
    // B tile 32x64 bf16
    {
      int i = tid;  // exactly 256 ops
      int k = i >> 3, no = (i & 7) * 8;
      async_ld_b128((unsigned)(uintptr_t)&Bs[k][no],
                    Bm + (size_t)(kb + k) * ldb + n0 + no);
    }
    if (kb + 32 < K) {
      __builtin_prefetch(&A[(size_t)(m0 + (tid >> 1)) * lda + kb + 32], 0, 3);
      __builtin_prefetch(&Bm[(size_t)(kb + 32 + (tid >> 3)) * ldb + n0], 0, 3);
    }
    wait_async0();
    __syncthreads();

    v16bf a0 = load_frag_a_bf(&As[wm * 32][0], 40);
    v16bf a1 = load_frag_a_bf(&As[wm * 32 + 16][0], 40);
    v16bf b0 = load_frag_b_bf(&Bs[0][wn * 32], 72);
    v16bf b1 = load_frag_b_bf(&Bs[0][wn * 32 + 16], 72);
    acc[0][0] = wmma_bf16(a0, b0, acc[0][0]);
    acc[0][1] = wmma_bf16(a0, b1, acc[0][1]);
    acc[1][0] = wmma_bf16(a1, b0, acc[1][0]);
    acc[1][1] = wmma_bf16(a1, b1, acc[1][1]);
  }

  const int nn = lane & 15;
  const int mb = (lane >> 4) * 8;
#pragma unroll
  for (int i = 0; i < 2; ++i)
#pragma unroll
    for (int j = 0; j < 2; ++j)
#pragma unroll
      for (int jj = 0; jj < 8; ++jj) {
        int gm = m0 + wm * 32 + i * 16 + mb + jj;
        int gn = n0 + wn * 32 + j * 16 + nn;
        float v = acc[i][j][jj];
        if (bias0) v += bias0[gn];
        if (bias1) v += bias1[gn];
        if (Cin) {
          if (CIN_BF16) v += (float)((const bf16*)Cin)[(size_t)gm * ldcin + gn];
          else          v += ((const float*)Cin)[(size_t)gm * ldcin + gn];
        }
        if (OUT_BF16) ((bf16*)Cout)[(size_t)gm * ldcout + gn] = (bf16)v;
        else          ((float*)Cout)[(size_t)gm * ldcout + gn] = v;
      }
}

// ---------------- LSTM gate kernel ----------------
__global__ void lstm_gates_kernel(const float* __restrict__ g,
                                  float* __restrict__ c,
                                  bf16* __restrict__ hbf,
                                  float* __restrict__ hs_cat,
                                  int t, int off) {
  int idx = blockIdx.x * blockDim.x + threadIdx.x;   // B*H
  int b = idx >> 8;
  int j = idx & 255;
  const float* gr = g + (size_t)b * G4H;
  float ig = 1.f / (1.f + expf(-gr[j]));
  float fg = 1.f / (1.f + expf(-gr[H_SZ + j]));
  float gg = tanhf(gr[2 * H_SZ + j]);
  float og = 1.f / (1.f + expf(-gr[3 * H_SZ + j]));
  float cc = fg * c[idx] + ig * gg;
  c[idx] = cc;
  float h = og * tanhf(cc);
  hbf[idx] = (bf16)h;
  hs_cat[((size_t)b * T_SZ + t) * CAT_F + off + j] = h;
}

// ---------------- PTP GEMM (fp32 WMMA 16x16x4, TDM-staged weights) ----------------
// Z[n,b,r,o] = sum_d x[b,n,d] * F[n,r,d,o]; d=0 const column folded in epilogue.
// Each 64-wide N block lives inside a single rank r, so the weight tile is a
// plain 2D slab (16 rows x 64 floats, row stride O) -> one TDM descriptor.
__global__ __launch_bounds__(256)
void ptp_gemm_kernel(const float* __restrict__ X, int ldx, int winstride,
                     const float* __restrict__ factor,
                     float* __restrict__ Z,
                     int D, int O, int shiftO, int Bt) {
  __shared__ float Xs[128][20];  // row stride 80B (16B aligned)
  __shared__ float Fs[16][68];   // row stride 272B (16B aligned); 64 data + 4 pad
  const int tid  = threadIdx.x;
  const int lane = tid & 31;
  const int w    = tid >> 5;
  const int wm   = w & 3;
  const int wn   = w >> 2;
  const int m0   = blockIdx.y * 128;
  const int n0   = blockIdx.x * 64;
  const int nwin = blockIdx.z;
  const float* Xb = X + (size_t)nwin * winstride;
  const float* Fb = factor + (size_t)nwin * R_RK * D * O;
  const int rblk = n0 >> shiftO;        // single rank per 64-wide block
  const int o0   = n0 & (O - 1);

  v8f acc[2][2] = {};
  const int Kmain = D - 1;

  for (int kb = 0; kb < Kmain; kb += 16) {
    __syncthreads();
    // X tile 128x16 fp32, 4 floats (16B) per async op
#pragma unroll
    for (int i = tid; i < 512; i += 256) {
      int m = i >> 2, kq = (i & 3) * 4;
      async_ld_b128((unsigned)(uintptr_t)&Xs[m][kq],
                    Xb + (size_t)(m0 + m) * ldx + kb + kq);
    }
    // F tile 16x64 fp32, row stride O
    const float* ftile = Fb + ((size_t)rblk * D + (1 + kb)) * O + o0;
#if defined(HAVE_TDM)
    if (w == 0) {   // one wave issues the DMA (EXEC ignored by tensor ops)
      unsigned long long ga = (unsigned long long)(uintptr_t)ftile;
      u32x4 g0;
      g0.x = 1u;                                            // count=1
      g0.y = (unsigned)(uintptr_t)&Fs[0][0];                // lds_addr
      g0.z = (unsigned)ga;                                  // global_addr lo
      g0.w = (unsigned)((ga >> 32) & 0x01FFFFFFull) | (2u << 30);  // hi + type=2
      i32x8 g1;
      g1[0] = (2 << 16) | (1 << 20) | (5 << 22) | (3 << 25); // 4B, pad on, 64DW/4DW
      g1[1] = (64 << 16);                                    // tensor_dim0 = 64
      g1[2] = (16 << 16);                                    // tensor_dim1 = 16
      g1[3] = (64 << 16);                                    // tile_dim0 = 64
      g1[4] = 16;                                            // tile_dim1 = 16
      g1[5] = O;                                             // dim0 stride lo32
      g1[6] = 0;
      g1[7] = 0;
      i32x4 gz = {0, 0, 0, 0};
#  if defined(TDM_SIX_ARG)
      i32x8 gp = {0, 0, 0, 0, 0, 0, 0, 0};
      __builtin_amdgcn_tensor_load_to_lds(g0, g1, gz, gz, gp, 0);
#  else
      __builtin_amdgcn_tensor_load_to_lds(g0, g1, gz, gz, 0);
#  endif
      wait_tensor0();
    }
#else
    // fallback: async-LDS staging of the weight slab
    {
      int i = tid;   // exactly 256 ops of 16B
      int k = i >> 4, jq = (i & 15) * 4;
      async_ld_b128((unsigned)(uintptr_t)&Fs[k][jq], ftile + (size_t)k * O + jq);
    }
#endif
    if (kb + 16 < Kmain)
      __builtin_prefetch(ftile + (size_t)(tid >> 4) * O + 16 * O, 0, 3);
    wait_async0();
    __syncthreads();

#pragma unroll
    for (int ks = 0; ks < 16; ks += 4) {
      v2f a0 = load_frag_a_f32(&Xs[wm * 32][ks], 20);
      v2f a1 = load_frag_a_f32(&Xs[wm * 32 + 16][ks], 20);
      v2f b0 = load_frag_b_f32(&Fs[ks][wn * 32], 68);
      v2f b1 = load_frag_b_f32(&Fs[ks][wn * 32 + 16], 68);
      acc[0][0] = wmma_f32(a0, b0, acc[0][0]);
      acc[0][1] = wmma_f32(a0, b1, acc[0][1]);
      acc[1][0] = wmma_f32(a1, b0, acc[1][0]);
      acc[1][1] = wmma_f32(a1, b1, acc[1][1]);
    }
  }

  const int nn = lane & 15;
  const int mb = (lane >> 4) * 8;
#pragma unroll
  for (int i = 0; i < 2; ++i)
#pragma unroll
    for (int j = 0; j < 2; ++j)
#pragma unroll
      for (int jj = 0; jj < 8; ++jj) {
        int gm = m0 + wm * 32 + i * 16 + mb + jj;
        int jg = n0 + wn * 32 + j * 16 + nn;
        int r = jg >> shiftO, o = jg & (O - 1);
        float f0 = Fb[(size_t)r * D * O + o];          // d == 0 column (bias 0.5)
        float v = acc[i][j][jj] + 0.5f * f0;
        Z[(((size_t)nwin * Bt + gm) * R_RK + r) * O + o] = v;
      }
}

// ---------------- PTP post: z^3, sign-sqrt, L2 norm, rank contraction ----------------
__global__ void ptp_post_kernel(const float* __restrict__ Z,
                                const float* __restrict__ weight,
                                const float* __restrict__ bias,
                                float* __restrict__ out,
                                int nW, int O, int Bt) {
  __shared__ float s[128];
  int b = blockIdx.x, n = blockIdx.y, o = threadIdx.x;
  float accum = 0.f;
  for (int r = 0; r < R_RK; ++r) {
    float z = Z[(((size_t)n * Bt + b) * R_RK + r) * O + o];
    float p = z * z * z;
    float ap = sqrtf(fabsf(p));
    p = (p < 0.f) ? -ap : ap;
    s[o] = p * p;
    __syncthreads();
    for (int st = O >> 1; st > 0; st >>= 1) {
      if (o < st) s[o] += s[o + st];
      __syncthreads();
    }
    float nrm = sqrtf(s[0]);
    __syncthreads();
    accum += weight[n * R_RK + r] * (p / nrm);
  }
  out[((size_t)b * nW + n) * O + o] = accum + bias[n * O + o];
}

// ---------------- host side ----------------
extern "C" void kernel_launch(void* const* d_in, const int* in_sizes, int n_in,
                              void* d_out, int out_size, void* d_ws, size_t ws_size,
                              hipStream_t stream) {
  (void)in_sizes; (void)n_in; (void)out_size; (void)ws_size;

  const float* audio_x  = (const float*)d_in[0];
  const float* video_x  = (const float*)d_in[1];
  const float* text_x   = (const float*)d_in[2];
  const float* aw_ih    = (const float*)d_in[3];
  const float* aw_hh    = (const float*)d_in[4];
  const float* ab_ih    = (const float*)d_in[5];
  const float* ab_hh    = (const float*)d_in[6];
  const float* vw_ih    = (const float*)d_in[7];
  const float* vw_hh    = (const float*)d_in[8];
  const float* vb_ih    = (const float*)d_in[9];
  const float* vb_hh    = (const float*)d_in[10];
  const float* txw_ih   = (const float*)d_in[11];
  const float* txw_hh   = (const float*)d_in[12];
  const float* txb_ih   = (const float*)d_in[13];
  const float* txb_hh   = (const float*)d_in[14];
  const float* l1_factor = (const float*)d_in[15];
  const float* l1_weight = (const float*)d_in[16];
  const float* l1_bias   = (const float*)d_in[17];
  const float* l2_factor = (const float*)d_in[18];
  const float* l2_weight = (const float*)d_in[19];
  const float* l2_bias   = (const float*)d_in[20];

  char* wp = (char*)d_ws;
  auto carve = [&](size_t bytes) -> void* {
    void* p = (void*)wp;
    wp += (bytes + 255) & ~(size_t)255;
    return p;
  };
  const int BT = B_SZ * T_SZ;
  bf16* xbf_a = (bf16*)carve((size_t)BT * A_IN * 2);
  bf16* xbf_v = (bf16*)carve((size_t)BT * V_IN * 2);
  bf16* xbf_t = (bf16*)carve((size_t)BT * TX_IN * 2);
  bf16* wT_a  = (bf16*)carve((size_t)A_IN * G4H * 2);
  bf16* wT_v  = (bf16*)carve((size_t)V_IN * G4H * 2);
  bf16* wT_t  = (bf16*)carve((size_t)TX_IN * G4H * 2);
  bf16* whhT_a = (bf16*)carve((size_t)H_SZ * G4H * 2);
  bf16* whhT_v = (bf16*)carve((size_t)H_SZ * G4H * 2);
  bf16* whhT_t = (bf16*)carve((size_t)H_SZ * G4H * 2);
  bf16* xp_a  = (bf16*)carve((size_t)BT * G4H * 2);
  bf16* xp_v  = (bf16*)carve((size_t)BT * G4H * 2);
  bf16* xp_t  = (bf16*)carve((size_t)BT * G4H * 2);
  float* hs_cat = (float*)carve((size_t)BT * CAT_F * 4);
  float* cbuf  = (float*)carve((size_t)3 * B_SZ * H_SZ * 4);
  bf16*  hbf   = (bf16*)carve((size_t)3 * B_SZ * H_SZ * 2);
  float* gbuf_a = (float*)carve((size_t)B_SZ * G4H * 4);
  float* gbuf_v = (float*)carve((size_t)B_SZ * G4H * 4);
  float* gbuf_t = (float*)carve((size_t)B_SZ * G4H * 4);
  float* z1 = (float*)carve((size_t)NW_L1 * B_SZ * R_RK * O0_SZ * 4);
  float* f1 = (float*)carve((size_t)B_SZ * NW_L1 * O0_SZ * 4);
  float* z2 = (float*)carve((size_t)B_SZ * R_RK * O1_SZ * 4);

  {
    int nwords = (3 * B_SZ * H_SZ * 4) / 4;
    zero_u32_kernel<<<(nwords + 255) / 256, 256, 0, stream>>>((unsigned int*)cbuf, nwords);
    int nwords2 = (3 * B_SZ * H_SZ * 2) / 4;
    zero_u32_kernel<<<(nwords2 + 255) / 256, 256, 0, stream>>>((unsigned int*)hbf, nwords2);
  }

  auto cvt = [&](const float* s, bf16* d, int n) {
    cvt_bf16_kernel<<<(n + 255) / 256, 256, 0, stream>>>(s, d, n);
  };
  auto cvtT = [&](const float* s, bf16* d, int cols) {
    int n = G4H * cols;
    cvt_t_bf16_kernel<<<(n + 255) / 256, 256, 0, stream>>>(s, d, G4H, cols);
  };
  cvt(audio_x, xbf_a, BT * A_IN);
  cvt(video_x, xbf_v, BT * V_IN);
  cvt(text_x,  xbf_t, BT * TX_IN);
  cvtT(aw_ih,  wT_a,  A_IN);
  cvtT(vw_ih,  wT_v,  V_IN);
  cvtT(txw_ih, wT_t,  TX_IN);
  cvtT(aw_hh,  whhT_a, H_SZ);
  cvtT(vw_hh,  whhT_v, H_SZ);
  cvtT(txw_hh, whhT_t, H_SZ);

  struct Mod {
    const bf16* xbf; int in; const bf16* wT; const bf16* whhT;
    const float* bih; const float* bhh; bf16* xp; bf16* hb; float* c; float* g; int off;
  };
  Mod mods[3] = {
    { xbf_t, TX_IN, wT_t, whhT_t, txb_ih, txb_hh, xp_t, hbf + 2 * B_SZ * H_SZ,
      cbuf + 2 * B_SZ * H_SZ, gbuf_t, 0 },
    { xbf_a, A_IN,  wT_a, whhT_a, ab_ih,  ab_hh,  xp_a, hbf,
      cbuf, gbuf_a, H_SZ },
    { xbf_v, V_IN,  wT_v, whhT_v, vb_ih,  vb_hh,  xp_v, hbf + B_SZ * H_SZ,
      cbuf + B_SZ * H_SZ, gbuf_v, 2 * H_SZ },
  };
  for (int m = 0; m < 3; ++m) {
    dim3 grid(G4H / 64, BT / 128);
    gemm_bf16_kernel<false, true><<<grid, 256, 0, stream>>>(
        mods[m].xbf, mods[m].in, mods[m].wT, G4H,
        nullptr, 0, mods[m].bih, mods[m].bhh,
        (void*)mods[m].xp, G4H, mods[m].in);
  }

  for (int t = 0; t < T_SZ; ++t) {
    for (int m = 0; m < 3; ++m) {
      dim3 grid(G4H / 64, B_SZ / 128);
      gemm_bf16_kernel<true, false><<<grid, 256, 0, stream>>>(
          mods[m].hb, H_SZ, mods[m].whhT, G4H,
          (const void*)(mods[m].xp + (size_t)t * G4H), T_SZ * G4H,
          nullptr, nullptr,
          (void*)mods[m].g, G4H, H_SZ);
    }
    for (int m = 0; m < 3; ++m) {
      lstm_gates_kernel<<<(B_SZ * H_SZ) / 256, 256, 0, stream>>>(
          mods[m].g, mods[m].c, mods[m].hb, hs_cat, t, mods[m].off);
    }
  }

  {
    dim3 grid((R_RK * O0_SZ) / 64, B_SZ / 128, NW_L1);
    ptp_gemm_kernel<<<grid, 256, 0, stream>>>(
        hs_cat, T_SZ * CAT_F, TW0 * CAT_F,
        l1_factor, z1, D1_SZ, O0_SZ, 7, B_SZ);
    dim3 pgrid(B_SZ, NW_L1);
    ptp_post_kernel<<<pgrid, O0_SZ, 0, stream>>>(
        z1, l1_weight, l1_bias, f1, NW_L1, O0_SZ, B_SZ);
  }

  {
    dim3 grid((R_RK * O1_SZ) / 64, B_SZ / 128, 1);
    ptp_gemm_kernel<<<grid, 256, 0, stream>>>(
        f1, NW_L1 * O0_SZ, 0,
        l2_factor, z2, D2_SZ, O1_SZ, 6, B_SZ);
    dim3 pgrid(B_SZ, 1);
    ptp_post_kernel<<<pgrid, O1_SZ, 0, stream>>>(
        z2, l2_weight, l2_bias, (float*)d_out, 1, O1_SZ, B_SZ);
  }
}
